// Transformer_26585847562611
// MI455X (gfx1250) — compile-verified, matrix operand order checked by reading, not run on hardware
//
#include <hip/hip_runtime.h>
#include <math.h>

#define B_   8
#define S_   512
#define D_   1024
#define H_   16
#define DH_  64
#define MS_  (B_*S_)     // 4096 rows
#define LAYER 1          // reference loop is not chained: only last layer matters

typedef unsigned short u16;
typedef __bf16 bf16_t;
typedef bf16_t v16bf __attribute__((ext_vector_type(16)));
typedef float  v8f   __attribute__((ext_vector_type(8)));

#define WMMA_BF16(a,b,c) __builtin_amdgcn_wmma_f32_16x16x32_bf16( \
    false,(a),false,(b),(short)0,(c),false,false)

static __device__ __forceinline__ u16 f2bf(float f){
  union { float f; unsigned u; } v; v.f = f;
  unsigned r = v.u + 0x7FFFu + ((v.u >> 16) & 1u); // RNE
  return (u16)(r >> 16);
}

// Load one 16x32 (or 32x16) bf16 fragment line for this lane.
// ISA 7.12.2: lane L holds row/col L%16; K-half = L/16; elements 0..7 hold
// K = half*8..half*8+7, elements 8..15 hold K = 16+half*8 .. 16+half*8+7.
static __device__ __forceinline__ v16bf fragload(const u16* row, int hl){
  union { uint4 q[2]; v16bf v; } f;
  f.q[0] = *(const uint4*)(row + hl*8);
  f.q[1] = *(const uint4*)(row + 16 + hl*8);
  return f.v;
}

// CDNA5 async Global->LDS copy (16B per lane), tracked by ASYNCcnt.
// For LDS-aperture generic addresses, addr[31:0] is the LDS address (ISA 10.2),
// so truncating the generic pointer yields the VDST LDS-offset operand.
static __device__ __forceinline__ void async_ld_b128(const u16* gptr, u16* lptr){
  unsigned loff = (unsigned)(unsigned long long)(uintptr_t)lptr;
  asm volatile("global_load_async_to_lds_b128 %0, %1, off"
               :: "v"(loff), "v"(gptr) : "memory");
}
static __device__ __forceinline__ void wait_async0(){
  asm volatile("s_wait_asynccnt 0x0" ::: "memory");
}

// ---------------------------------------------------------------------------
// Elementwise converts
// ---------------------------------------------------------------------------
__global__ __launch_bounds__(256) void cvt_f32_bf16_kernel(
    const float* __restrict__ src, u16* __restrict__ dst, int n, float scale)
{
  int i = blockIdx.x * 256 + threadIdx.x;
  if (i < n) dst[i] = f2bf(src[i] * scale);
}

__global__ __launch_bounds__(256) void gelu_cvt_kernel(
    const float* __restrict__ src, u16* __restrict__ dst, int n)
{
  int i = blockIdx.x * 256 + threadIdx.x;
  if (i < n) {
    float x = src[i];
    float g = 0.5f * x * (1.0f + erff(x * 0.70710678118654752f)); // exact GELU
    dst[i] = f2bf(g);
  }
}

// v [B,S,D] bf16 -> vt [B,H,DH,S] bf16
__global__ __launch_bounds__(256) void transpose_v_kernel(
    const u16* __restrict__ v, u16* __restrict__ vt)
{
  int i = blockIdx.x * 256 + threadIdx.x;            // over B*H*DH*S
  int s  = i & (S_ - 1);
  int t  = i >> 9;                                   // /S_
  int d  = t & (DH_ - 1);
  int t2 = t >> 6;                                   // /DH_
  int h  = t2 & (H_ - 1);
  int b  = t2 >> 4;                                  // /H_
  vt[i] = v[((size_t)(b*S_ + s))*D_ + h*DH_ + d];
}

// ---------------------------------------------------------------------------
// Input LN (unbiased std, eps on std) + sinusoidal positional encoding
// ---------------------------------------------------------------------------
__global__ __launch_bounds__(256) void ln_pe_kernel(
    const float* __restrict__ feat, const float* __restrict__ g,
    const float* __restrict__ beta, float* __restrict__ xo, u16* __restrict__ xb)
{
  __shared__ float r1[256], r2[256];
  const int row = blockIdx.x, tid = threadIdx.x;
  const int pos = row % S_;
  float v[4]; float s = 0.f, s2 = 0.f;
  #pragma unroll
  for (int i = 0; i < 4; ++i) {
    int c = tid + i*256;
    v[i] = feat[(size_t)row*D_ + c];
    s += v[i]; s2 += v[i]*v[i];
  }
  r1[tid] = s; r2[tid] = s2; __syncthreads();
  for (int off = 128; off > 0; off >>= 1) {
    if (tid < off) { r1[tid] += r1[tid+off]; r2[tid] += r2[tid+off]; }
    __syncthreads();
  }
  float mean = r1[0] * (1.f/D_);
  float var  = (r2[0] - (float)D_*mean*mean) * (1.f/(D_-1));   // ddof=1
  float den  = sqrtf(fmaxf(var, 0.f)) + 1e-6f;                 // eps on std
  #pragma unroll
  for (int i = 0; i < 4; ++i) {
    int c = tid + i*256;
    float xn  = g[c]*(v[i]-mean)/den + beta[c];
    float ang = (float)pos * powf(10000.f, -(2.f*(float)c)/(float)D_);
    float pe  = (c & 1) ? cosf(ang) : sinf(ang);
    float val = xn + pe;
    xo[(size_t)row*D_ + c] = val;
    xb[(size_t)row*D_ + c] = f2bf(val);
  }
}

// LN(a + resid) -> out_f32 (+ optional bf16 copy)
__global__ __launch_bounds__(256) void add_ln_kernel(
    const float* __restrict__ a, const float* __restrict__ resid,
    const float* __restrict__ g, const float* __restrict__ beta,
    float* __restrict__ out, u16* __restrict__ outb)
{
  __shared__ float r1[256], r2[256];
  const int row = blockIdx.x, tid = threadIdx.x;
  float v[4]; float s = 0.f, s2 = 0.f;
  #pragma unroll
  for (int i = 0; i < 4; ++i) {
    int c = tid + i*256;
    v[i] = a[(size_t)row*D_ + c] + resid[(size_t)row*D_ + c];
    s += v[i]; s2 += v[i]*v[i];
  }
  r1[tid] = s; r2[tid] = s2; __syncthreads();
  for (int off = 128; off > 0; off >>= 1) {
    if (tid < off) { r1[tid] += r1[tid+off]; r2[tid] += r2[tid+off]; }
    __syncthreads();
  }
  float mean = r1[0] * (1.f/D_);
  float var  = (r2[0] - (float)D_*mean*mean) * (1.f/(D_-1));
  float den  = sqrtf(fmaxf(var, 0.f)) + 1e-6f;
  #pragma unroll
  for (int i = 0; i < 4; ++i) {
    int c = tid + i*256;
    float val = g[c]*(v[i]-mean)/den + beta[c];
    out[(size_t)row*D_ + c] = val;
    if (outb) outb[(size_t)row*D_ + c] = f2bf(val);
  }
}

// ---------------------------------------------------------------------------
// bf16 WMMA GEMM: C[M,N] = A[M,K] * W[N,K]^T + bias[N]   (fp32 out)
// block tile 128x64, K-step 32, 8 wave32s each doing 32x32.
// Double-buffered LDS fed by CDNA5 async Global->LDS copies (ASYNCcnt):
// tile k+1 streams into the other buffer while WMMAs consume tile k.
// ---------------------------------------------------------------------------
#define BM 128
#define BN 64
#define BK 32
#define LDS_STR 40   // 32 + 8 halves pad -> 80B rows, conflict-free, 16B aligned

__global__ __launch_bounds__(256) void gemm_bf16_wmma(
    const u16* __restrict__ A, const u16* __restrict__ W,
    const float* __restrict__ bias, float* __restrict__ C,
    int M, int N, int K)
{
  __shared__ u16 As[2][BM*LDS_STR];
  __shared__ u16 Bs[2][BN*LDS_STR];
  const int tid = threadIdx.x, lane = tid & 31, wid = tid >> 5;
  const int wm = wid & 3, wn = wid >> 2;
  const int m0 = blockIdx.y * BM, n0 = blockIdx.x * BN;
  const int r = lane & 15, hl = lane >> 4;

  const v8f z8 = {0.f,0.f,0.f,0.f,0.f,0.f,0.f,0.f};
  v8f acc[2][2]; acc[0][0]=z8; acc[0][1]=z8; acc[1][0]=z8; acc[1][1]=z8;

  // per-thread chunk coordinates for cooperative tile loads
  const int arow0 = tid >> 2,            ac8 = (tid & 3) << 3;   // A chunk 0
  const int arow1 = (tid + 256) >> 2;                            // A chunk 1
  const int brow  = tid >> 2,            bc8 = (tid & 3) << 3;   // B chunk

  // issue async loads for one K-tile into LDS buffer `bi`
  auto issue_tile = [&](int k0, int bi) {
    async_ld_b128(A + (size_t)(m0+arow0)*K + k0 + ac8, &As[bi][arow0*LDS_STR + ac8]);
    async_ld_b128(A + (size_t)(m0+arow1)*K + k0 + ac8, &As[bi][arow1*LDS_STR + ac8]);
    async_ld_b128(W + (size_t)(n0+brow )*K + k0 + bc8, &Bs[bi][brow *LDS_STR + bc8]);
  };

  issue_tile(0, 0);
  const int NK = K / BK;
  for (int kt = 0; kt < NK; ++kt) {
    const int cur = kt & 1;
    wait_async0();        // my async stores to LDS landed
    __syncthreads();      // everyone's landed; everyone done reading buf `cur^1`
    if (kt + 1 < NK) issue_tile((kt + 1) * BK, cur ^ 1);

    const u16* a_lds = As[cur];
    const u16* b_lds = Bs[cur];
    v16bf af[2], bfrag[2];
    #pragma unroll
    for (int mi = 0; mi < 2; ++mi)
      af[mi] = fragload(a_lds + (wm*32 + mi*16 + r)*LDS_STR, hl);
    #pragma unroll
    for (int ni = 0; ni < 2; ++ni)
      bfrag[ni] = fragload(b_lds + (wn*32 + ni*16 + r)*LDS_STR, hl);

    #pragma unroll
    for (int mi = 0; mi < 2; ++mi)
      #pragma unroll
      for (int ni = 0; ni < 2; ++ni)
        acc[mi][ni] = WMMA_BF16(af[mi], bfrag[ni], acc[mi][ni]);
  }

  // C/D layout: lane holds col = lane%16; vgpr rr holds row rr + 8*(lane/16)
  #pragma unroll
  for (int mi = 0; mi < 2; ++mi)
    #pragma unroll
    for (int ni = 0; ni < 2; ++ni) {
      int gn = n0 + wn*32 + ni*16 + r;
      float bv = bias[gn];
      int gm = m0 + wm*32 + mi*16 + hl*8;
      #pragma unroll
      for (int rr = 0; rr < 8; ++rr)
        C[(size_t)(gm+rr)*N + gn] = acc[mi][ni][rr] + bv;
    }
}

// ---------------------------------------------------------------------------
// Attention: one block per (b,h,q-tile of 32). Full 32x512 score panel in LDS,
// masked softmax, bf16 P, P*Vt with WMMA. Q arrives pre-scaled by 1/8.
// ---------------------------------------------------------------------------
#define QT  32
#define SCS 516   // fp32 score row stride (pad 4): conflict-free column access
#define PBS 520   // bf16 prob row stride (pad 8): 1040B rows, 16B aligned

__global__ __launch_bounds__(256) void attn_wmma(
    const u16* __restrict__ q, const u16* __restrict__ k,
    const u16* __restrict__ vt, const int* __restrict__ mask,
    float* __restrict__ ctx)
{
  extern __shared__ char smem[];
  float* sc = (float*)smem;                                  // [QT][SCS]
  u16*   pb = (u16*)(smem + (size_t)QT*SCS*sizeof(float));   // [QT][PBS]

  const int tid = threadIdx.x, lane = tid & 31, wid = tid >> 5;
  const int r = lane & 15, hl = lane >> 4;
  const int b = blockIdx.x / H_, h = blockIdx.x % H_;
  const int q0 = blockIdx.y * QT;
  const v8f z8 = {0.f,0.f,0.f,0.f,0.f,0.f,0.f,0.f};

  const u16* Qp = q + (size_t)b*S_*D_ + h*DH_;
  const u16* Kp = k + (size_t)b*S_*D_ + h*DH_;

  // ---- scores: each wave covers 2 m-tiles x 4 n-tiles (kpos blocks) ----
  v16bf af[2][2];
  #pragma unroll
  for (int mi = 0; mi < 2; ++mi)
    #pragma unroll
    for (int ks = 0; ks < 2; ++ks)
      af[mi][ks] = fragload(Qp + (size_t)(q0 + mi*16 + r)*D_ + ks*32, hl);

  #pragma unroll
  for (int nt = 0; nt < 4; ++nt) {
    int n0 = (wid*4 + nt)*16;
    v16bf kf[2];
    #pragma unroll
    for (int ks = 0; ks < 2; ++ks)
      kf[ks] = fragload(Kp + (size_t)(n0 + r)*D_ + ks*32, hl);
    #pragma unroll
    for (int mi = 0; mi < 2; ++mi) {
      v8f acc = z8;
      acc = WMMA_BF16(af[mi][0], kf[0], acc);
      acc = WMMA_BF16(af[mi][1], kf[1], acc);
      #pragma unroll
      for (int rr = 0; rr < 8; ++rr)
        sc[(mi*16 + hl*8 + rr)*SCS + n0 + r] = acc[rr];
    }
  }
  __syncthreads();

  // ---- masked softmax, one thread per row ----
  if (tid < QT) {
    float* row = sc + tid*SCS;
    const int* mrow = mask + b*S_;
    float mx = -3.4e38f;
    for (int j = 0; j < S_; ++j) {
      float v = row[j];
      if (mrow[j] == 0) v = -1e18f;
      row[j] = v; mx = fmaxf(mx, v);
    }
    float sum = 0.f;
    for (int j = 0; j < S_; ++j) { float e = __expf(row[j]-mx); row[j] = e; sum += e; }
    float inv = 1.f / sum;
    u16* prow = pb + tid*PBS;
    for (int j = 0; j < S_; ++j) prow[j] = f2bf(row[j]*inv);
  }
  __syncthreads();

  // ---- ctx = P[32x512] * Vt^T (Vt rows are K-contiguous) ----
  const u16* Vp = vt + (size_t)(b*H_ + h)*DH_*S_;
  const int mi = wid & 1, ni = wid >> 1;
  v8f acc = z8;
  for (int ks = 0; ks < 16; ++ks) {
    int k0 = ks*32;
    v16bf a  = fragload(pb + (mi*16 + r)*PBS + k0, hl);
    v16bf bb = fragload(Vp + (size_t)(ni*16 + r)*S_ + k0, hl);
    acc = WMMA_BF16(a, bb, acc);
  }
  #pragma unroll
  for (int rr = 0; rr < 8; ++rr)
    ctx[((size_t)(b*S_ + q0 + mi*16 + hl*8 + rr))*D_ + h*DH_ + ni*16 + r] = acc[rr];
}

// ---------------------------------------------------------------------------
extern "C" void kernel_launch(void* const* d_in, const int* in_sizes, int n_in,
                              void* d_out, int out_size, void* d_ws, size_t ws_size,
                              hipStream_t stream) {
  (void)in_sizes; (void)n_in; (void)out_size; (void)ws_size;
  const float* features = (const float*)d_in[0];
  const int*   mask     = (const int*)  d_in[1];
  const float* ln_in_g  = (const float*)d_in[2];
  const float* ln_in_b  = (const float*)d_in[3];
  const size_t WOF = (size_t)LAYER * D_ * D_, BOF = (size_t)LAYER * D_;
  const float* Wq = (const float*)d_in[4]  + WOF; const float* bq = (const float*)d_in[5]  + BOF;
  const float* Wk = (const float*)d_in[6]  + WOF; const float* bk = (const float*)d_in[7]  + BOF;
  const float* Wv = (const float*)d_in[8]  + WOF; const float* bv = (const float*)d_in[9]  + BOF;
  const float* Wo = (const float*)d_in[10] + WOF; const float* bo = (const float*)d_in[11] + BOF;
  const float* l1g = (const float*)d_in[12] + BOF; const float* l1b = (const float*)d_in[13] + BOF;
  const float* W1 = (const float*)d_in[14] + WOF; const float* b1 = (const float*)d_in[15] + BOF;
  const float* W2 = (const float*)d_in[16] + WOF; const float* b2 = (const float*)d_in[17] + BOF;
  const float* l2g = (const float*)d_in[18] + BOF; const float* l2b = (const float*)d_in[19] + BOF;
  float* out = (float*)d_out;

  // workspace carve-up
  char* w = (char*)d_ws;
  auto alloc = [&](size_t bytes) { void* p = w; w += (bytes + 255) & ~(size_t)255; return p; };
  const size_t ND = (size_t)MS_ * D_;           // 4M elems
  float* x_f32   = (float*)alloc(ND*4);
  float* tmp_f32 = (float*)alloc(ND*4);
  float* ctx_f32 = (float*)alloc(ND*4);
  float* s_f32   = (float*)alloc(ND*4);
  u16* x_bf   = (u16*)alloc(ND*2);
  u16* q_bf   = (u16*)alloc(ND*2);
  u16* k_bf   = (u16*)alloc(ND*2);
  u16* v_bf   = (u16*)alloc(ND*2);
  u16* vt_bf  = (u16*)alloc(ND*2);
  u16* ctx_bf = (u16*)alloc(ND*2);
  u16* s_bf   = (u16*)alloc(ND*2);
  u16* h_bf   = (u16*)alloc(ND*2);
  u16* wq_bf  = (u16*)alloc((size_t)D_*D_*2);
  u16* wk_bf  = (u16*)alloc((size_t)D_*D_*2);
  u16* wv_bf  = (u16*)alloc((size_t)D_*D_*2);
  u16* wo_bf  = (u16*)alloc((size_t)D_*D_*2);
  u16* w1_bf  = (u16*)alloc((size_t)D_*D_*2);
  u16* w2_bf  = (u16*)alloc((size_t)D_*D_*2);

  const int NW = D_*D_, GW = NW/256;            // weight convert grid
  const int NA = (int)ND, GA = NA/256;          // activation grid
  dim3 gemm_grid(D_/BN, MS_/BM);                // 16 x 32
  dim3 attn_grid(B_*H_, S_/QT);                 // 128 x 16
  const size_t ATT_SMEM = (size_t)QT*SCS*4 + (size_t)QT*PBS*2;  // 99328 B

  // weights -> bf16
  cvt_f32_bf16_kernel<<<GW,256,0,stream>>>(Wq, wq_bf, NW, 1.f);
  cvt_f32_bf16_kernel<<<GW,256,0,stream>>>(Wk, wk_bf, NW, 1.f);
  cvt_f32_bf16_kernel<<<GW,256,0,stream>>>(Wv, wv_bf, NW, 1.f);
  cvt_f32_bf16_kernel<<<GW,256,0,stream>>>(Wo, wo_bf, NW, 1.f);
  cvt_f32_bf16_kernel<<<GW,256,0,stream>>>(W1, w1_bf, NW, 1.f);
  cvt_f32_bf16_kernel<<<GW,256,0,stream>>>(W2, w2_bf, NW, 1.f);

  // x = LN(features) + PE
  ln_pe_kernel<<<MS_,256,0,stream>>>(features, ln_in_g, ln_in_b, x_f32, x_bf);

  // Q (pre-scaled by 1/sqrt(DH)=1/8), K, V
  gemm_bf16_wmma<<<gemm_grid,256,0,stream>>>(x_bf, wq_bf, bq, tmp_f32, MS_, D_, D_);
  cvt_f32_bf16_kernel<<<GA,256,0,stream>>>(tmp_f32, q_bf, NA, 0.125f);
  gemm_bf16_wmma<<<gemm_grid,256,0,stream>>>(x_bf, wk_bf, bk, tmp_f32, MS_, D_, D_);
  cvt_f32_bf16_kernel<<<GA,256,0,stream>>>(tmp_f32, k_bf, NA, 1.f);
  gemm_bf16_wmma<<<gemm_grid,256,0,stream>>>(x_bf, wv_bf, bv, tmp_f32, MS_, D_, D_);
  cvt_f32_bf16_kernel<<<GA,256,0,stream>>>(tmp_f32, v_bf, NA, 1.f);
  transpose_v_kernel<<<GA,256,0,stream>>>(v_bf, vt_bf);

  // attention
  attn_wmma<<<attn_grid,256,ATT_SMEM,stream>>>(q_bf, k_bf, vt_bf, mask, ctx_f32);
  cvt_f32_bf16_kernel<<<GA,256,0,stream>>>(ctx_f32, ctx_bf, NA, 1.f);

  // output projection + residual LN1
  gemm_bf16_wmma<<<gemm_grid,256,0,stream>>>(ctx_bf, wo_bf, bo, tmp_f32, MS_, D_, D_);
  add_ln_kernel<<<MS_,256,0,stream>>>(tmp_f32, x_f32, l1g, l1b, s_f32, s_bf);

  // FFN
  gemm_bf16_wmma<<<gemm_grid,256,0,stream>>>(s_bf, w1_bf, b1, tmp_f32, MS_, D_, D_);
  gelu_cvt_kernel<<<GA,256,0,stream>>>(tmp_f32, h_bf, NA);
  gemm_bf16_wmma<<<gemm_grid,256,0,stream>>>(h_bf, w2_bf, b2, tmp_f32, MS_, D_, D_);

  // final residual LN2 -> output
  add_ln_kernel<<<MS_,256,0,stream>>>(tmp_f32, s_f32, l2g, l2b, out, nullptr);
}
